// TopLevelVQVAE_39977555591194
// MI455X (gfx1250) — matrix operand order repaired and checked by reading, not compiled, not run on hardware
//
#include <hip/hip_runtime.h>
#include <hip/hip_bf16.h>
#include <stdint.h>
#include <stddef.h>

typedef __attribute__((ext_vector_type(16))) __bf16 v16bf;
typedef __attribute__((ext_vector_type(8)))  float  v8f;

#define EPSN 1e-5f

// ---- problem sizes -------------------------------------------------------
#define NB     32
#define CIN    3
#define H0     256      // input spatial
#define H1D    128      // after conv1
#define C1     16
#define H2D    64       // after conv2
#define ZD     64       // latent dim
#define KCB    512      // codewords per stage
#define NSTG   6        // 2 blocks * 3 stages
#define NROWS  (NB*H2D*H2D)          // 131072 latent vectors
#define N_H1   (NB*C1*H1D*H1D)       // 8388608
#define N_FLAT (NROWS*ZD)            // 8388608
#define N_OUT  (NB*CIN*H0*H0)        // 6291456

// ---- workspace byte offsets (256B aligned) -------------------------------
#define OFF_H1   ((size_t)0)             // 33554432 B  (h1 / d1 reuse)
#define OFF_FLAT ((size_t)33554432)      // 33554432 B  (enc in [N,Z] NHWC)
#define OFF_R    ((size_t)67108864)      // 33554432 B  (residual [N,Z])
#define OFF_CB16 ((size_t)100663296)     //   393216 B  (bf16 codebooks)
#define OFF_CBN  ((size_t)101056512)     //    12288 B  (-0.5*||e||^2 f32)
#define OFF_BN   ((size_t)101068800)     //      256 B  (sum16,sumsq16,mean16,rstd16)
#define OFF_IST  ((size_t)101069056)     //    16384 B  (instance-norm mean/rstd)

// padded LDS codebook row: 72 bf16 = 144 B (9x16B). Lane start-bank stride
// 36 dwords mod 64 -> 16 distinct banks (mod-4 aligned) -> conflict-free b128.
#define PADB 72
#define RVQ_LDS_BYTES (KCB*PADB*2 + KCB*4)   // 73728 + 2048 = 75776 (dynamic)

// ==========================================================================
// zero accumulators (loss slot + BN sum/sumsq)
__global__ void k_zero(float* loss, float* bn) {
    if (threadIdx.x == 0) *loss = 0.0f;
    if (threadIdx.x < 32) bn[threadIdx.x] = 0.0f;
}

// f32 codebooks -> bf16 copy + pre-scaled norms (-0.5*||e||^2)
__global__ void k_cvt_cb(const float* __restrict__ cb, __bf16* __restrict__ cb16,
                         float* __restrict__ cbn) {
    int e = blockIdx.x * blockDim.x + threadIdx.x;     // 0..3071 codewords
    if (e >= NSTG * KCB) return;
    const float* src = cb + (size_t)e * ZD;
    float s = 0.f;
    for (int d = 0; d < ZD; ++d) {
        float f = src[d];
        cb16[(size_t)e * ZD + d] = (__bf16)f;
        s += f * f;
    }
    cbn[e] = -0.5f * s;
}

// ---- conv1: 3->16, k4 s2 p1, NCHW out ------------------------------------
__global__ __launch_bounds__(256)
void k_conv1(const float* __restrict__ x, const float* __restrict__ w,
             const float* __restrict__ b, float* __restrict__ out) {
    int i = blockIdx.x * blockDim.x + threadIdx.x;
    if (i >= N_H1) return;
    int ox = i & 127, oy = (i >> 7) & 127, co = (i >> 14) & 15, n = i >> 18;
    float acc = b[co];
    #pragma unroll
    for (int ky = 0; ky < 4; ++ky) {
        int iy = 2 * oy - 1 + ky;
        if (iy < 0 || iy >= H0) continue;
        #pragma unroll
        for (int kx = 0; kx < 4; ++kx) {
            int ix = 2 * ox - 1 + kx;
            if (ix < 0 || ix >= H0) continue;
            #pragma unroll
            for (int ci = 0; ci < 3; ++ci)
                acc += x[(((size_t)n * 3 + ci) * H0 + iy) * H0 + ix] *
                       w[((co * 3 + ci) * 4 + ky) * 4 + kx];
        }
    }
    out[i] = acc;
}

// ---- instance-norm stats, NCHW: one block per (n,c) ----------------------
__global__ __launch_bounds__(256)
void k_in_stats_nchw(const float* __restrict__ x, float* __restrict__ st, int HW) {
    __shared__ float s1[256], s2[256];
    const float* base = x + (size_t)blockIdx.x * HW;
    float a = 0.f, b = 0.f;
    for (int p = threadIdx.x; p < HW; p += 256) { float v = base[p]; a += v; b += v * v; }
    s1[threadIdx.x] = a; s2[threadIdx.x] = b;
    __syncthreads();
    for (int o = 128; o > 0; o >>= 1) {
        if (threadIdx.x < o) { s1[threadIdx.x] += s1[threadIdx.x + o]; s2[threadIdx.x] += s2[threadIdx.x + o]; }
        __syncthreads();
    }
    if (threadIdx.x == 0) {
        float m = s1[0] / (float)HW;
        float v = s2[0] / (float)HW - m * m;
        st[2 * blockIdx.x] = m;
        st[2 * blockIdx.x + 1] = rsqrtf(v + EPSN);
    }
}

__global__ __launch_bounds__(256)
void k_in_apply_nchw(float* __restrict__ x, const float* __restrict__ st,
                     int HW, int total) {
    int i = blockIdx.x * blockDim.x + threadIdx.x;
    if (i >= total) return;
    int nc = i / HW;
    float v = (x[i] - st[2 * nc]) * st[2 * nc + 1];
    x[i] = v > 0.f ? v : 0.f;
}

// ---- conv2: 16->64, k4 s2 p1, writes NHWC "flat" [N,Z] -------------------
__global__ __launch_bounds__(256)
void k_conv2(const float* __restrict__ h, const float* __restrict__ w,
             const float* __restrict__ b, float* __restrict__ flat) {
    int i = blockIdx.x * blockDim.x + threadIdx.x;     // i == flat index
    if (i >= N_FLAT) return;
    int co = i & 63, ox = (i >> 6) & 63, oy = (i >> 12) & 63, n = i >> 18;
    float acc = b[co];
    #pragma unroll
    for (int ky = 0; ky < 4; ++ky) {
        int iy = 2 * oy - 1 + ky;
        if (iy < 0 || iy >= H1D) continue;
        #pragma unroll
        for (int kx = 0; kx < 4; ++kx) {
            int ix = 2 * ox - 1 + kx;
            if (ix < 0 || ix >= H1D) continue;
            #pragma unroll
            for (int ci = 0; ci < 16; ++ci)
                acc += h[(((size_t)n * 16 + ci) * H1D + iy) * H1D + ix] *
                       w[((co * 16 + ci) * 4 + ky) * 4 + kx];
        }
    }
    flat[i] = acc;
}

// ---- instance-norm stats on NHWC flat: block per (n,c) = 2048 blocks -----
__global__ __launch_bounds__(256)
void k_in_stats_nhwc(const float* __restrict__ flat, float* __restrict__ st) {
    __shared__ float s1[256], s2[256];
    int n = blockIdx.x >> 6, c = blockIdx.x & 63;
    const float* base = flat + ((size_t)n * 4096) * ZD + c;
    float a = 0.f, b = 0.f;
    for (int p = threadIdx.x; p < 4096; p += 256) { float v = base[(size_t)p * ZD]; a += v; b += v * v; }
    s1[threadIdx.x] = a; s2[threadIdx.x] = b;
    __syncthreads();
    for (int o = 128; o > 0; o >>= 1) {
        if (threadIdx.x < o) { s1[threadIdx.x] += s1[threadIdx.x + o]; s2[threadIdx.x] += s2[threadIdx.x + o]; }
        __syncthreads();
    }
    if (threadIdx.x == 0) {
        float m = s1[0] / 4096.f;
        float v = s2[0] / 4096.f - m * m;
        st[2 * blockIdx.x] = m;
        st[2 * blockIdx.x + 1] = rsqrtf(v + EPSN);
    }
}

// apply + relu; also seed residual r = enc
__global__ __launch_bounds__(256)
void k_in_apply_nhwc(float* __restrict__ flat, float* __restrict__ r,
                     const float* __restrict__ st) {
    int i = blockIdx.x * blockDim.x + threadIdx.x;
    if (i >= N_FLAT) return;
    int c = i & 63, n = i >> 18;
    int nc = n * 64 + c;
    float v = (flat[i] - st[2 * nc]) * st[2 * nc + 1];
    v = v > 0.f ? v : 0.f;
    flat[i] = v;
    r[i] = v;
}

// ==========================================================================
// Fused RVQ stage: bf16 WMMA distance GEMM + argmax + residual update + loss.
// dist = ||e||^2 - 2 r.e = -2*( -0.5||e||^2 + r.e ): init acc with -0.5||e||^2,
// argmin(dist) == argmax(acc). One wave owns 32 rows (two M-blocks sharing the
// same B fragments); workgroup = 8 waves = 256 rows. Codebook in padded,
// bank-conflict-free dynamic LDS.
__global__ __launch_bounds__(256)
void k_rvq_stage(float* __restrict__ r, const __bf16* __restrict__ cb16,
                 const float* __restrict__ cbn, const float* __restrict__ cbf32,
                 float* __restrict__ loss) {
    extern __shared__ char smem[];
    __bf16* cbL  = (__bf16*)smem;                        // 512 rows * 144 B
    float*  nrmL = (float*)(smem + KCB * PADB * 2);      // 2 KB
    {   // cooperative copy into padded LDS rows (32 dwords data / 36 dword stride)
        const uint32_t* src = (const uint32_t*)cb16;
        uint32_t* dst = (uint32_t*)cbL;
        for (int i = threadIdx.x; i < KCB * ZD / 2; i += 256) {
            int row = i >> 5, off = i & 31;
            dst[row * (PADB / 2) + off] = src[i];
        }
        for (int i = threadIdx.x; i < KCB; i += 256) nrmL[i] = cbn[i];
    }
    __syncthreads();

    const int lane = threadIdx.x & 31;
    const int wave = threadIdx.x >> 5;
    const int rowBase = blockIdx.x * 256 + wave * 32;

    // ---- A fragments: 2 blocks of 16 rows x K=64 (two 16x32 bf16 frags each)
    // A k order used here (k=16h+s): lanes 0-15 M=0..15 K 0..15, lanes 16-31 K 16..31; VGPR v -> K=2v(+khalf)
    const int mA  = lane & 15;
    const int khA = (lane >> 4) << 4;                    // 0 or 16
    const float* rrow0 = r + (size_t)(rowBase + mA) * ZD;
    const float* rrow1 = rrow0 + 16 * ZD;
    v16bf a0, a1, a2, a3;
    #pragma unroll
    for (int v = 0; v < 8; ++v) {
        int kb = 2 * v + khA;
        float2 p0 = *(const float2*)(rrow0 + kb);
        float2 p1 = *(const float2*)(rrow0 + kb + 32);
        float2 p2 = *(const float2*)(rrow1 + kb);
        float2 p3 = *(const float2*)(rrow1 + kb + 32);
        a0[2 * v] = (__bf16)p0.x; a0[2 * v + 1] = (__bf16)p0.y;
        a1[2 * v] = (__bf16)p1.x; a1[2 * v + 1] = (__bf16)p1.y;
        a2[2 * v] = (__bf16)p2.x; a2[2 * v + 1] = (__bf16)p2.y;
        a3[2 * v] = (__bf16)p3.x; a3[2 * v + 1] = (__bf16)p3.y;
    }

    // B k order here (as A): lanes 0-15 -> K 0..15 (pairs per VGPR), lanes 16-31 -> K 16..31; N = lane&15
    const int nB  = lane & 15;
    const int khB = (lane >> 4) << 4;                    // 0 or 16

    float maxv[16]; int maxi[16];                        // [0..7]=rows 0-15, [8..15]=rows 16-31
    #pragma unroll
    for (int v = 0; v < 16; ++v) { maxv[v] = -3.0e38f; maxi[v] = 0; }

    #pragma unroll 2
    for (int nt = 0; nt < KCB / 16; ++nt) {              // 32 column tiles
        const int col = nt * 16 + nB;
        const __bf16* crow = cbL + col * PADB;           // padded row
        v16bf b0, b1;
        #pragma unroll
        for (int v = 0; v < 8; ++v) {
            b0[2 * v]     = crow[khB + 2 * v];
            b0[2 * v + 1] = crow[khB + 2 * v + 1];
            b1[2 * v]     = crow[khB + 2 * v + 32];
            b1[2 * v + 1] = crow[khB + 2 * v + 33];
        }
        float nv = nrmL[col];                            // -0.5*||e||^2
        v8f accA, accB;
        #pragma unroll
        for (int v = 0; v < 8; ++v) { accA[v] = nv; accB[v] = nv; }
        accA = __builtin_amdgcn_wmma_f32_16x16x32_bf16(false, a0, false, b0,
                                                       (short)0, accA, false, false);
        accA = __builtin_amdgcn_wmma_f32_16x16x32_bf16(false, a1, false, b1,
                                                       (short)0, accA, false, false);
        accB = __builtin_amdgcn_wmma_f32_16x16x32_bf16(false, a2, false, b0,
                                                       (short)0, accB, false, false);
        accB = __builtin_amdgcn_wmma_f32_16x16x32_bf16(false, a3, false, b1,
                                                       (short)0, accB, false, false);
        // score = -0.5*dist; per-lane col ascends, strict > keeps first index
        #pragma unroll
        for (int v = 0; v < 8; ++v) {
            if (accA[v] > maxv[v])     { maxv[v]     = accA[v]; maxi[v]     = col; }
            if (accB[v] > maxv[8 + v]) { maxv[8 + v] = accB[v]; maxi[8 + v] = col; }
        }
    }

    // reduce over N within each 16-lane half; tie-break: smaller index
    #pragma unroll
    for (int off = 8; off >= 1; off >>= 1) {
        #pragma unroll
        for (int v = 0; v < 16; ++v) {
            float ov = __shfl_xor(maxv[v], off, 32);
            int   oi = __shfl_xor(maxi[v], off, 32);
            if (ov > maxv[v] || (ov == maxv[v] && oi < maxi[v])) { maxv[v] = ov; maxi[v] = oi; }
        }
    }

    // residual update (f32 codebook) + loss = 1.25 * sum(r_new^2) / (N*D)
    float sq = 0.0f;
    #pragma unroll
    for (int m = 0; m < 32; ++m) {
        int g = m >> 4, ml = m & 15;                     // block, row-in-block
        int idx = __shfl(maxi[g * 8 + (ml & 7)], (ml < 8) ? 0 : 16, 32);
        float* rr = r + (size_t)(rowBase + m) * ZD;
        const float* q = cbf32 + (size_t)idx * ZD;
        int d = lane * 2;
        float2 rv = *(float2*)(rr + d);
        float2 qv = *(const float2*)(q + d);
        rv.x -= qv.x; rv.y -= qv.y;
        *(float2*)(rr + d) = rv;
        sq += rv.x * rv.x + rv.y * rv.y;
    }
    #pragma unroll
    for (int off = 16; off >= 1; off >>= 1) sq += __shfl_xor(sq, off, 32);
    if (lane == 0)
        atomicAdd(loss, 1.25f * sq * (1.0f / ((float)NROWS * (float)ZD)));
}

// ==========================================================================
// dec_in = flat - r (in place into flat; halves deconv1 load traffic)
__global__ __launch_bounds__(256)
void k_sub(float* __restrict__ flat, const float* __restrict__ r) {
    int i = blockIdx.x * blockDim.x + threadIdx.x;
    if (i < N_FLAT) flat[i] -= r[i];
}

// deconv1: ConvTranspose 64->16, k4 s2 p1; input = dec_in (NHWC flat)
__global__ __launch_bounds__(256)
void k_deconv1(const float* __restrict__ din, const float* __restrict__ w,
               const float* __restrict__ b, float* __restrict__ out) {
    int i = blockIdx.x * blockDim.x + threadIdx.x;
    if (i >= N_H1) return;
    int ox = i & 127, oy = (i >> 7) & 127, co = (i >> 14) & 15, n = i >> 18;
    float acc = b[co];
    #pragma unroll
    for (int ky = 0; ky < 4; ++ky) {
        int ty = oy + ky - 2;
        if (ty < 0 || (ty & 1) || (ty >> 1) >= H2D) continue;
        int iy = ty >> 1;
        #pragma unroll
        for (int kx = 0; kx < 4; ++kx) {
            int tx = ox + kx - 2;
            if (tx < 0 || (tx & 1) || (tx >> 1) >= H2D) continue;
            int ix = tx >> 1;
            size_t base = ((size_t)n * 4096 + iy * 64 + ix) * ZD;
            #pragma unroll 8
            for (int ci = 0; ci < ZD; ++ci)
                acc += din[base + ci] * w[((co * ZD + ci) * 4 + ky) * 4 + kx];
        }
    }
    out[i] = acc;
}

// ---- batch-norm over (N,H,W) per channel: atomic partial sums ------------
__global__ __launch_bounds__(256)
void k_bn_stats(const float* __restrict__ x, float* __restrict__ bn) {
    __shared__ float s1[256], s2[256];
    int c = blockIdx.x, n = blockIdx.y;
    const float* base = x + ((size_t)(n * 16 + c)) * 16384;
    float a = 0.f, b = 0.f;
    for (int p = threadIdx.x; p < 16384; p += 256) { float v = base[p]; a += v; b += v * v; }
    s1[threadIdx.x] = a; s2[threadIdx.x] = b;
    __syncthreads();
    for (int o = 128; o > 0; o >>= 1) {
        if (threadIdx.x < o) { s1[threadIdx.x] += s1[threadIdx.x + o]; s2[threadIdx.x] += s2[threadIdx.x + o]; }
        __syncthreads();
    }
    if (threadIdx.x == 0) { atomicAdd(&bn[c], s1[0]); atomicAdd(&bn[16 + c], s2[0]); }
}

__global__ void k_bn_finalize(float* bn) {
    int c = threadIdx.x;
    if (c >= 16) return;
    const float cnt = 32.0f * 16384.0f;
    float m = bn[c] / cnt;
    float v = bn[16 + c] / cnt - m * m;
    bn[32 + c] = m;
    bn[48 + c] = rsqrtf(v + EPSN);
}

__global__ __launch_bounds__(256)
void k_bn_apply(float* __restrict__ x, const float* __restrict__ bn,
                const float* __restrict__ g, const float* __restrict__ be) {
    int i = blockIdx.x * blockDim.x + threadIdx.x;
    if (i >= N_H1) return;
    int c = (i >> 14) & 15;
    x[i] = (x[i] - bn[32 + c]) * bn[48 + c] * g[c] + be[c];
}

// ---- deconv2: ConvTranspose 16->3, k4 s2 p1 -> decoded output ------------
__global__ __launch_bounds__(256)
void k_deconv2(const float* __restrict__ d1, const float* __restrict__ w,
               const float* __restrict__ b, float* __restrict__ out) {
    int i = blockIdx.x * blockDim.x + threadIdx.x;
    if (i >= N_OUT) return;
    int ox = i % 256;
    int oy = (i / 256) % 256;
    int co = (i / 65536) % 3;
    int n  = i / 196608;
    float acc = b[co];
    #pragma unroll
    for (int ky = 0; ky < 4; ++ky) {
        int ty = oy + ky - 2;
        if (ty < 0 || (ty & 1) || (ty >> 1) >= H1D) continue;
        int iy = ty >> 1;
        #pragma unroll
        for (int kx = 0; kx < 4; ++kx) {
            int tx = ox + kx - 2;
            if (tx < 0 || (tx & 1) || (tx >> 1) >= H1D) continue;
            int ix = tx >> 1;
            #pragma unroll
            for (int ci = 0; ci < 16; ++ci)
                acc += d1[(((size_t)n * 16 + ci) * H1D + iy) * H1D + ix] *
                       w[((co * 16 + ci) * 4 + ky) * 4 + kx];
        }
    }
    out[i] = acc;
}

// ==========================================================================
extern "C" void kernel_launch(void* const* d_in, const int* in_sizes, int n_in,
                              void* d_out, int out_size, void* d_ws, size_t ws_size,
                              hipStream_t stream) {
    const float* x     = (const float*)d_in[0];
    const float* ew1   = (const float*)d_in[1];
    const float* eb1   = (const float*)d_in[2];
    const float* ew2   = (const float*)d_in[3];
    const float* eb2   = (const float*)d_in[4];
    const float* cb    = (const float*)d_in[5];
    const float* dw1   = (const float*)d_in[6];
    const float* db1   = (const float*)d_in[7];
    const float* gam   = (const float*)d_in[8];
    const float* bet   = (const float*)d_in[9];
    const float* dw2   = (const float*)d_in[10];
    const float* db2   = (const float*)d_in[11];

    char* ws = (char*)d_ws;
    float*  h1   = (float*)(ws + OFF_H1);     // conv1 out, reused as deconv1 out
    float*  flat = (float*)(ws + OFF_FLAT);   // encoded latents [N,Z], later dec_in
    float*  rres = (float*)(ws + OFF_R);      // residual [N,Z]
    __bf16* cb16 = (__bf16*)(ws + OFF_CB16);
    float*  cbn  = (float*)(ws + OFF_CBN);
    float*  bn   = (float*)(ws + OFF_BN);
    float*  ist  = (float*)(ws + OFF_IST);

    float* dec  = (float*)d_out;
    float* loss = dec + N_OUT;

    k_zero<<<1, 64, 0, stream>>>(loss, bn);
    k_cvt_cb<<<(NSTG * KCB + 255) / 256, 256, 0, stream>>>(cb, cb16, cbn);

    // encoder
    k_conv1<<<N_H1 / 256, 256, 0, stream>>>(x, ew1, eb1, h1);
    k_in_stats_nchw<<<NB * C1, 256, 0, stream>>>(h1, ist, H1D * H1D);
    k_in_apply_nchw<<<N_H1 / 256, 256, 0, stream>>>(h1, ist, H1D * H1D, N_H1);
    k_conv2<<<N_FLAT / 256, 256, 0, stream>>>(h1, ew2, eb2, flat);
    k_in_stats_nhwc<<<NB * ZD, 256, 0, stream>>>(flat, ist);
    k_in_apply_nhwc<<<N_FLAT / 256, 256, 0, stream>>>(flat, rres, ist);

    // residual VQ: 6 stages, WMMA distance GEMM fused with argmax/update/loss
    for (int s = 0; s < NSTG; ++s)
        k_rvq_stage<<<NROWS / 256, 256, RVQ_LDS_BYTES, stream>>>(
            rres, cb16 + (size_t)s * KCB * ZD, cbn + s * KCB,
            cb + (size_t)s * KCB * ZD, loss);

    // decoder
    k_sub<<<N_FLAT / 256, 256, 0, stream>>>(flat, rres);       // dec_in = flat - r
    k_deconv1<<<N_H1 / 256, 256, 0, stream>>>(flat, dw1, db1, h1);
    k_bn_stats<<<dim3(16, 32), 256, 0, stream>>>(h1, bn);
    k_bn_finalize<<<1, 16, 0, stream>>>(bn);
    k_bn_apply<<<N_H1 / 256, 256, 0, stream>>>(h1, bn, gam, bet);
    k_deconv2<<<N_OUT / 256, 256, 0, stream>>>(h1, dw2, db2, dec);
}
